// AutoCorrelation_39393440039209
// MI455X (gfx1250) — matrix operand order repaired and checked by reading, not compiled, not run on hardware
//
#include <hip/hip_runtime.h>
#include <hip/hip_bf16.h>
#include <math.h>

#define D_MODEL 1024
#define N_HEADS 16
#define DH      64
#define BB      8
#define LL      2048
#define KTOP    7

typedef __attribute__((ext_vector_type(16))) __bf16 v16bf;
typedef __attribute__((ext_vector_type(8)))  float  v8f;
typedef __attribute__((ext_vector_type(4)))  int    i4vec;

#if __has_builtin(__builtin_amdgcn_global_load_async_to_lds_b128)
#define HAVE_ASYNC_LDS 1
#endif

#define GAS128(p) ((__attribute__((address_space(1))) i4vec*)(p))
#define LAS128(p) ((__attribute__((address_space(3))) i4vec*)(p))

__device__ __forceinline__ unsigned bfbits(float f) {
  union { float f; unsigned u; } v; v.f = f;
  return (v.u + 0x7FFFu + ((v.u >> 16) & 1u)) >> 16;   // RNE
}
__device__ __forceinline__ __bf16 f2bf(float f) {
  unsigned short h = (unsigned short)bfbits(f);
  return __builtin_bit_cast(__bf16, h);
}
// packed f32x2 -> bf16x2 (v_cvt_pk_bf16_f32 when available)
__device__ __forceinline__ unsigned pkbf(float x, float y) {
#if __has_builtin(__builtin_amdgcn_cvt_pk_bf16_f32)
  auto pk = __builtin_amdgcn_cvt_pk_bf16_f32(x, y);
  return __builtin_bit_cast(unsigned, pk);
#else
  return bfbits(x) | (bfbits(y) << 16);
#endif
}

struct u4x2 { uint4 a, b; };
__device__ __forceinline__ v16bf ld_frag(const __bf16* p0, const __bf16* p1) {
  u4x2 t;
  t.a = *reinterpret_cast<const uint4*>(p0);
  t.b = *reinterpret_cast<const uint4*>(p1);
  return __builtin_bit_cast(v16bf, t);
}

__device__ __forceinline__ void wait_async0() {
#ifdef HAVE_ASYNC_LDS
#if __has_builtin(__builtin_amdgcn_s_wait_asynccnt)
  __builtin_amdgcn_s_wait_asynccnt(0);
#else
  asm volatile("s_wait_asynccnt 0x0" ::: "memory");
#endif
#endif
}

// ---------------------------------------------------------------------------
// Weight prep: f32 (K x N) -> bf16 transposed (N x K, k-contiguous).
// ---------------------------------------------------------------------------
__global__ __launch_bounds__(256)
void transpose_to_bf16(const float* __restrict__ in, __bf16* __restrict__ out)
{
  __shared__ float tile[32][33];
  const int k0 = blockIdx.x * 32, n0 = blockIdx.y * 32;
  const int c = threadIdx.x & 31, r0 = threadIdx.x >> 5;   // 8 rows per pass
  #pragma unroll
  for (int rr = 0; rr < 32; rr += 8)
    tile[r0 + rr][c] = in[(size_t)(k0 + r0 + rr) * D_MODEL + n0 + c];
  __syncthreads();
  #pragma unroll
  for (int rr = 0; rr < 32; rr += 8)
    out[(size_t)(n0 + r0 + rr) * D_MODEL + k0 + c] = f2bf(tile[c][r0 + rr]);
}

// ---------------------------------------------------------------------------
// bf16 WMMA GEMM:  C[M,N] = A[M,K] (f32 -> bf16) * W (bf16, pre-transposed NxK)
// 256 threads (8 waves), tile 128(M) x 64(N), K-step 32, double-buffered LDS.
// B tile staged with global_load_async_to_lds_b128 overlapped with WMMA.
// ---------------------------------------------------------------------------
template<bool TSTORE>
__global__ __launch_bounds__(256)
void gemm_bf16(const float* __restrict__ A, const __bf16* __restrict__ Wt,
               float* __restrict__ out, int M, int N, int K, int Ltime)
{
  __shared__ __align__(16) __bf16 As[2][128 * 32];   // row-major, k contiguous
  __shared__ __align__(16) __bf16 Bs[2][64 * 32];    // n-major,  k contiguous

  const int tid  = threadIdx.x;
  const int lane = tid & 31;
  const int wid  = tid >> 5;
  const int half = lane >> 4;
  const int l16  = lane & 15;
  const int m0   = blockIdx.x * 128;
  const int n0   = blockIdx.y * 64;

  v8f acc[4];
  #pragma unroll
  for (int nt = 0; nt < 4; ++nt)
    #pragma unroll
    for (int e = 0; e < 8; ++e) acc[nt][e] = 0.0f;

  const int ar0 = tid >> 3;            // 0..31
  const int ac4 = (tid & 7) * 4;       // 0..28 (float4 col group)
  const int bn  = tid >> 2;            // 0..63 (B tile row = n)
  const int bkc = (tid & 3) * 8;       // 0..24 (8 bf16 = 16B chunk of k)

  auto stage = [&](int k0, int buf) {
    // B tile 64(n) x 32(k): one async b128 per thread
    {
      const __bf16* gsrc = Wt + (size_t)(n0 + bn) * K + k0 + bkc;
      __bf16* ldst = &Bs[buf][bn * 32 + bkc];
#ifdef HAVE_ASYNC_LDS
      __builtin_amdgcn_global_load_async_to_lds_b128(GAS128(gsrc), LAS128(ldst),
                                                     0, 0);
#else
      *reinterpret_cast<uint4*>(ldst) = *reinterpret_cast<const uint4*>(gsrc);
#endif
    }
    // A tile 128(m) x 32(k): f32 -> packed bf16 pairs (v_cvt_pk_bf16_f32)
    #pragma unroll
    for (int rr = 0; rr < 4; ++rr) {
      int r = ar0 + rr * 32;
      const float4 av = *reinterpret_cast<const float4*>(
          A + (size_t)(m0 + r) * K + k0 + ac4);
      uint2 pp;
      pp.x = pkbf(av.x, av.y);
      pp.y = pkbf(av.z, av.w);
      *reinterpret_cast<uint2*>(
          reinterpret_cast<char*>(As[buf]) + (size_t)(r * 32 + ac4) * 2) = pp;
    }
  };

  stage(0, 0);
  wait_async0();
  __syncthreads();

  const int KSTEPS = K / 32;
  for (int ks = 0; ks < KSTEPS; ++ks) {
    const int cur = ks & 1;
    if (ks + 1 < KSTEPS) {
      stage((ks + 1) * 32, cur ^ 1);               // overlap with WMMA below
      if (ks + 2 < KSTEPS)                         // global_prefetch_b8 ahead
        __builtin_prefetch(A + (size_t)(m0 + ar0) * K + (ks + 2) * 32 + ac4,
                           0, 1);
    }
    // fragments: two ds_load_b128 each
    const __bf16* arow = &As[cur][(wid * 16 + l16) * 32];
    const v16bf af = ld_frag(arow + half * 8, arow + 16 + half * 8);
    #pragma unroll
    for (int nt = 0; nt < 4; ++nt) {
      const __bf16* bcol = &Bs[cur][(nt * 16 + l16) * 32];
      const v16bf bfv = ld_frag(bcol + half * 8, bcol + 16 + half * 8);
      acc[nt] = __builtin_amdgcn_wmma_f32_16x16x32_bf16(
          false, af, false, bfv, (short)0, acc[nt], false, false);
    }
    wait_async0();     // next-buffer async writes landed
    __syncthreads();   // everyone done reading cur / writing next
  }

  if constexpr (TSTORE) {
    // transpose through LDS so the (channel, t) store is t-contiguous
    __shared__ float Cs[128 * 64];
    #pragma unroll
    for (int nt = 0; nt < 4; ++nt)
      #pragma unroll
      for (int r = 0; r < 8; ++r)
        Cs[(wid * 16 + half * 8 + r) * 64 + nt * 16 + l16] = acc[nt][r];
    __syncthreads();
    const int bbi = m0 / Ltime;   // M block never straddles a batch (128 | 2048)
    const int t0  = m0 % Ltime;
    for (int idx = tid; idx < 128 * 64; idx += 256) {
      int nl = idx >> 7;          // channel within tile
      int ml = idx & 127;         // time within tile (contiguous stores)
      out[((size_t)bbi * N + (n0 + nl)) * Ltime + (t0 + ml)] = Cs[ml * 64 + nl];
    }
  } else {
    #pragma unroll
    for (int nt = 0; nt < 4; ++nt)
      #pragma unroll
      for (int r = 0; r < 8; ++r)
        out[(size_t)(m0 + wid * 16 + half * 8 + r) * N + (n0 + nt * 16 + l16)]
            = acc[nt][r];
  }
}

// ---------------------------------------------------------------------------
// 2048-point Stockham radix-2 FFT in LDS (self-sorting, ping-pong buffers).
// 11 stages (odd) => result ends in (dr0, di0).
// ---------------------------------------------------------------------------
__device__ void fft2048(float* sr0, float* si0, float* dr0, float* di0,
                        float sign, int tid)
{
  float *sr = sr0, *si = si0, *dr = dr0, *di = di0;
  int n = 2048, log2s = 0;
  while (n > 1) {
    const int m = n >> 1;
    const float theta0 = sign * 6.283185307179586f / (float)n;
    const int s     = 1 << log2s;
    const int smask = s - 1;
    for (int j = tid; j < 1024; j += 256) {
      int p = j >> log2s;
      int q = j & smask;
      float cs, sn;
      __sincosf(theta0 * (float)p, &sn, &cs);
      float ar = sr[q + s * p],        ai = si[q + s * p];
      float br = sr[q + s * (p + m)],  bi = si[q + s * (p + m)];
      dr[q + s * (2 * p)]     = ar + br;
      di[q + s * (2 * p)]     = ai + bi;
      float tr = ar - br, ti = ai - bi;
      dr[q + s * (2 * p + 1)] = tr * cs - ti * sn;
      di[q + s * (2 * p + 1)] = tr * sn + ti * cs;
    }
    __syncthreads();
    float* t;
    t = sr; sr = dr; dr = t;
    t = si; si = di; di = t;
    n >>= 1; ++log2s;
  }
}

// ---------------------------------------------------------------------------
// Per-(b,h,d) lane: corr = irfft(rfft(Q) * conj(rfft(K))), top-7, softmax.
// ---------------------------------------------------------------------------
__global__ __launch_bounds__(256)
void autocorr_fft(const float* __restrict__ Qt, const float* __restrict__ Kt,
                  float* __restrict__ wOut, int* __restrict__ tauOut)
{
  __shared__ float xr[2048], xi[2048], yr[2048], yi[2048];
  __shared__ float sval[256];
  __shared__ int   sidx[256];
  __shared__ float topv[KTOP];
  __shared__ int   topi[KTOP];

  const int tid = threadIdx.x;
  const int row = blockIdx.x;                 // (b*H + h)*Dh + d
  const float* q = Qt + (size_t)row * LL;
  const float* k = Kt + (size_t)row * LL;

  for (int j = tid; j < LL; j += 256) { xr[j] = q[j]; xi[j] = k[j]; }
  __syncthreads();
  fft2048(xr, xi, yr, yi, -1.0f, tid);        // Z = FFT(Q + iK) in (yr, yi)

  for (int j = tid; j < LL; j += 256) {       // P = Qf * conj(Kf)
    int j2 = (LL - j) & (LL - 1);
    float ar = yr[j], ai = yi[j], br = yr[j2], bi = yi[j2];
    float Qr = 0.5f * (ar + br), Qi = 0.5f * (ai - bi);
    float Kr = 0.5f * (ai + bi), Ki = 0.5f * (br - ar);
    xr[j] = Qr * Kr + Qi * Ki;
    xi[j] = Qi * Kr - Qr * Ki;
  }
  __syncthreads();
  fft2048(xr, xi, yr, yi, +1.0f, tid);        // inverse; corr = yr / N
  for (int j = tid; j < LL; j += 256) yr[j] *= (1.0f / 2048.0f);
  __syncthreads();

  for (int pass = 0; pass < KTOP; ++pass) {   // top-7 via block argmax
    float best = -INFINITY; int bidx = 0;
    for (int j = tid; j < LL; j += 256) {
      float v = yr[j];
      if (v > best) { best = v; bidx = j; }
    }
    sval[tid] = best; sidx[tid] = bidx;
    __syncthreads();
    for (int off = 128; off > 0; off >>= 1) {
      if (tid < off && sval[tid + off] > sval[tid]) {
        sval[tid] = sval[tid + off]; sidx[tid] = sidx[tid + off];
      }
      __syncthreads();
    }
    if (tid == 0) {
      topv[pass] = sval[0]; topi[pass] = sidx[0];
      yr[sidx[0]] = -INFINITY;
    }
    __syncthreads();
  }

  if (tid == 0) {
    float mx = topv[0];
    for (int i = 1; i < KTOP; ++i) mx = fmaxf(mx, topv[i]);
    float den = 0.0f, e[KTOP];
    for (int i = 0; i < KTOP; ++i) { e[i] = __expf(topv[i] - mx); den += e[i]; }
    float inv = 1.0f / den;
    for (int i = 0; i < KTOP; ++i) {
      wOut[(size_t)row * 8 + i]   = e[i] * inv;
      tauOut[(size_t)row * 8 + i] = topi[i];
    }
  }
}

// ---------------------------------------------------------------------------
// out[b,h,d,t] = sum_i w_i * V[b,h,(d - tau_i) mod 64, t], written in the
// reference's scrambled flat order p = d*(H*L) + h*L + t.
// ---------------------------------------------------------------------------
__global__ __launch_bounds__(256)
void combine_roll(const float* __restrict__ Vt, const float* __restrict__ wIn,
                  const int* __restrict__ tauIn, float* __restrict__ mixed)
{
  const int row = blockIdx.x;          // (b*H + h)*64 + d
  const int b = row >> 10;
  const int h = (row >> 6) & 15;
  const int d = row & 63;

  float w[KTOP]; int src[KTOP];
  #pragma unroll
  for (int i = 0; i < KTOP; ++i) {
    w[i] = wIn[(size_t)row * 8 + i];
    int tau = tauIn[(size_t)row * 8 + i];
    int ds = (d - tau) % 64; if (ds < 0) ds += 64;
    src[i] = ds;
  }
  const size_t vbase = ((size_t)b * N_HEADS + h) * DH;
  float* outp = mixed + (size_t)b * ((size_t)D_MODEL * LL)
                      + (size_t)d * ((size_t)N_HEADS * LL)
                      + (size_t)h * LL;
  for (int t = threadIdx.x; t < LL; t += 256) {
    float acc = 0.0f;
    #pragma unroll
    for (int i = 0; i < KTOP; ++i)
      acc += w[i] * Vt[(vbase + src[i]) * LL + t];
    outp[t] = acc;
  }
}

// ---------------------------------------------------------------------------
extern "C" void kernel_launch(void* const* d_in, const int* in_sizes, int n_in,
                              void* d_out, int out_size, void* d_ws, size_t ws_size,
                              hipStream_t stream)
{
  const float* queries = (const float*)d_in[0];
  const float* keys    = (const float*)d_in[1];
  const float* values  = (const float*)d_in[2];
  const float* wq = (const float*)d_in[3];
  const float* wk = (const float*)d_in[4];
  const float* wv = (const float*)d_in[5];
  const float* wo = (const float*)d_in[6];
  float* out = (float*)d_out;

  char* ws = (char*)d_ws;
  const size_t szT = (size_t)BB * D_MODEL * LL * sizeof(float);   // 64 MiB
  float*  Qt  = (float*)(ws);
  float*  Kt  = (float*)(ws + szT);
  float*  Vt  = (float*)(ws + 2 * szT);
  __bf16* wqb = (__bf16*)(ws + 3 * szT);
  __bf16* wkb = wqb + (size_t)D_MODEL * D_MODEL;
  __bf16* wvb = wkb + (size_t)D_MODEL * D_MODEL;
  __bf16* wob = wvb + (size_t)D_MODEL * D_MODEL;
  float*  wW  = (float*)(wob + (size_t)D_MODEL * D_MODEL);
  int*    tauW = (int*)(wW + (size_t)BB * N_HEADS * DH * 8);
  float*  mixed = Qt;   // Qt dead after autocorr_fft; reuse for scrambled out

  const int M = BB * LL, N = D_MODEL, K = D_MODEL;
  dim3 gw(M / 128, N / 64), blk(256);
  dim3 gt(D_MODEL / 32, D_MODEL / 32);

  transpose_to_bf16<<<gt, blk, 0, stream>>>(wq, wqb);
  transpose_to_bf16<<<gt, blk, 0, stream>>>(wk, wkb);
  transpose_to_bf16<<<gt, blk, 0, stream>>>(wv, wvb);
  transpose_to_bf16<<<gt, blk, 0, stream>>>(wo, wob);

  gemm_bf16<true ><<<gw, blk, 0, stream>>>(queries, wqb, Qt, M, N, K, LL);
  gemm_bf16<true ><<<gw, blk, 0, stream>>>(keys,    wkb, Kt, M, N, K, LL);
  gemm_bf16<true ><<<gw, blk, 0, stream>>>(values,  wvb, Vt, M, N, K, LL);

  autocorr_fft<<<BB * N_HEADS * DH, 256, 0, stream>>>(Qt, Kt, wW, tauW);
  combine_roll<<<BB * N_HEADS * DH, 256, 0, stream>>>(Vt, wW, tauW, mixed);

  gemm_bf16<false><<<gw, blk, 0, stream>>>(mixed, wob, out, M, N, K, LL);
}
